// TSFN_Model_31215822307594
// MI455X (gfx1250) — matrix operand order repaired and checked by reading, not compiled, not run on hardware
//
#include <hip/hip_runtime.h>
#include <hip/hip_bf16.h>
#include <math.h>

typedef _Float16 h16;
typedef __attribute__((ext_vector_type(16))) _Float16 v16h;
typedef __attribute__((ext_vector_type(8)))  _Float16 v8h;
typedef __attribute__((ext_vector_type(8)))  float    v8f;
typedef __attribute__((ext_vector_type(4)))  unsigned int u32x4;
typedef __attribute__((ext_vector_type(8)))  int i32x8;
typedef __attribute__((ext_vector_type(4)))  int i32x4;

// Tensor Data Mover availability (5-arg on ROCm7.2/clang-22, 6-arg on clang-23+)
#ifndef USE_TDM
#if defined(__has_builtin)
#if __has_builtin(__builtin_amdgcn_tensor_load_to_lds)
#define USE_TDM 1
#endif
#endif
#endif
#ifndef USE_TDM
#define USE_TDM 0
#endif

// ---------------------------------------------------------------------------
// WMMA fragment helpers (wave32, 16x16x32 f16 -> f32)
// A (16x32): lane L (m=L&15): halves 0..7 = k{0..7}(+8 if L>=16),
//            halves 8..15 = k{16..23}(+8 if L>=16).
// B (32x16): lane L (n=L&15): same k order as A (k{0..7},k{16..23}, +8 if L>=16).
// C/D f32:   vgpr r -> m = r + (L<16?0:8), n = L&15.
// ---------------------------------------------------------------------------
static __device__ __forceinline__ v16h mk16(v8h lo, v8h hi) {
  v16h r;
#pragma unroll
  for (int i = 0; i < 8; ++i) { r[i] = lo[i]; r[i + 8] = hi[i]; }
  return r;
}

static __device__ __forceinline__ v16h load_a_frag_lds(const h16* sA, int stride,
                                                       int kkk, int lane) {
  const h16* p = sA + (lane & 15) * stride + kkk + ((lane < 16) ? 0 : 8);
  v8h lo = *(const v8h*)(p);
  v8h hi = *(const v8h*)(p + 16);
  return mk16(lo, hi);
}

static __device__ __forceinline__ v16h load_b_frag(const h16* W, size_t ldw,
                                                   int n0, int kk, int lane) {
  const h16* p = W + (size_t)(n0 + (lane & 15)) * ldw + kk + ((lane < 16) ? 0 : 8);
  v8h lo = *(const v8h*)(p);
  v8h hi = *(const v8h*)(p + 16);
  return mk16(lo, hi);
}

static __device__ __forceinline__ v8f wmma_f32(v16h a, v16h b, v8f c) {
  return __builtin_amdgcn_wmma_f32_16x16x32_f16(false, a, false, b, (short)0, c,
                                                false, false);
}

// ---------------------------------------------------------------------------
// TDM: one descriptor loads a [tile_h x tile_w] f16 tile (row stride g_stride
// elements) from global into LDS, rows packed contiguously (stride = tile_w).
// D# group0: count=1 | lds_addr | global_addr | type=2  (ISA 8.3)
// D# group1: data_size=2B, tensor_dim0=tile_w, tensor_dim1=tile_h,
//            tile_dim0=tile_w, tile_dim1=tile_h, dim0_stride=g_stride (ISA 8.4)
// ---------------------------------------------------------------------------
#if USE_TDM
static __device__ __forceinline__ void tdm_load_tile_f16(unsigned lds_off,
                                                         const h16* g, int tile_w,
                                                         int tile_h, int g_stride) {
  const unsigned long long ga = (unsigned long long)(size_t)g;
  u32x4 g0;
  g0[0] = 1u;  // count=1, no gather, user mode
  g0[1] = lds_off;
  g0[2] = (unsigned)(ga & 0xFFFFFFFFu);
  g0[3] = (unsigned)((ga >> 32) & 0x01FFFFFFu) | (2u << 30);  // type=2 (image)
  i32x8 g1;
  g1[0] = (int)(1u << 16);  // workgroup_mask=0, data_size=1 (2 bytes)
  g1[1] = (int)(((unsigned)tile_w & 0xFFFFu) << 16);              // tensor_dim0 lo
  g1[2] = (int)((((unsigned)tile_w >> 16) & 0xFFFFu) |
                (((unsigned)tile_h & 0xFFFFu) << 16));            // dim0 hi | dim1 lo
  g1[3] = (int)(((unsigned)tile_w & 0xFFFFu) << 16);              // dim1 hi=0 | tile_dim0
  g1[4] = (int)((unsigned)tile_h & 0xFFFFu);                      // tile_dim1, tile_dim2=0
  g1[5] = (int)(unsigned)g_stride;                                // tensor_dim0_stride lo
  g1[6] = 0;
  g1[7] = 0;
  const i32x4 z4 = {0, 0, 0, 0};
#if __clang_major__ >= 23
  const i32x8 z8 = {0, 0, 0, 0, 0, 0, 0, 0};
  __builtin_amdgcn_tensor_load_to_lds(g0, g1, z4, z4, z8, 0);
#else
  __builtin_amdgcn_tensor_load_to_lds(g0, g1, z4, z4, 0);
#endif
}
#endif

// Stage A tile [16 x chunk] (chunk multiple of 32) into LDS; ends with barrier.
static __device__ __forceinline__ void stage_a_tile(h16* sA, const h16* Ap,
                                                    size_t row0, int ldap, int kk,
                                                    int chunk, int tid, int wave) {
#if USE_TDM
  if (wave == 0) {
    tdm_load_tile_f16((unsigned)(size_t)sA, Ap + row0 * (size_t)ldap + kk, chunk,
                      16, ldap);
    __builtin_amdgcn_s_wait_tensorcnt(0);
  }
#else
  const int halfc = chunk >> 1;
  for (int u = tid; u < 8 * chunk; u += 256) {
    const int row = u / halfc;
    const int colp = (u - row * halfc) * 2;
    *(unsigned*)&sA[row * chunk + colp] =
        *(const unsigned*)&Ap[(row0 + row) * (size_t)ldap + kk + colp];
  }
#endif
  __syncthreads();
}

// ---------------------------------------------------------------------------
// Generic WMMA GEMM:  C[M,N] = A[M,Kp] * W[N,Kp]^T (+ A2*W2^T) + bias (+relu)
// Kp multiples of 32, M and N multiples of 16. grid = (M/16, ceil(N/128)),
// block = 256 (8 waves, one 16x16 tile per wave). A staged 16x128 per round
// via the Tensor Data Mover.
// ---------------------------------------------------------------------------
__global__ void __launch_bounds__(256) gemm_wmma_kernel(
    const h16* __restrict__ A, int lda, const h16* __restrict__ W, int Kp,
    const h16* __restrict__ A2, int lda2, const h16* __restrict__ W2, int K2p,
    const float* __restrict__ bias, float* __restrict__ C, int ldc,
    h16* __restrict__ C16, int ldc16, int M, int N, int do_relu) {
  __shared__ __align__(16) h16 sA[16 * 128];
  const int tid = threadIdx.x;
  const int lane = tid & 31;
  const int wave = tid >> 5;
  const int m0 = blockIdx.x * 16;
  const int n0 = blockIdx.y * 128 + wave * 16;
  const bool active = (n0 < N);

  v8f c = {};
#pragma unroll 1
  for (int pass = 0; pass < 2; ++pass) {
    const h16* Ap = pass ? A2 : A;
    const h16* Wp = pass ? W2 : W;
    const int Kpp = pass ? K2p : Kp;
    const int ldap = pass ? lda2 : lda;
    if (Ap == nullptr) continue;
    for (int kk = 0; kk < Kpp; kk += 128) {
      const int chunk = (Kpp - kk < 128) ? (Kpp - kk) : 128;
      if (kk + 128 < Kpp)  // global_prefetch_b8 on next A chunk
        __builtin_prefetch(&Ap[(size_t)(m0 + (tid >> 4)) * ldap + kk + 128 +
                               ((tid & 15) * 8)], 0, 1);
      __syncthreads();
      stage_a_tile(sA, Ap, (size_t)m0, ldap, kk, chunk, tid, wave);
      for (int kkk = 0; kkk < chunk; kkk += 32) {
        v16h a = load_a_frag_lds(sA, chunk, kkk, lane);
        if (active) {
          v16h b = load_b_frag(Wp, (size_t)Kpp, n0, kk + kkk, lane);
          c = wmma_f32(a, b, c);
        }
      }
    }
  }
  if (active) {
    const int n = n0 + (lane & 15);
    const float bv = bias ? bias[n] : 0.0f;
#pragma unroll
    for (int r = 0; r < 8; ++r) {
      const int m = m0 + r + ((lane < 16) ? 0 : 8);
      float v = c[r] + bv;
      if (do_relu) v = fmaxf(v, 0.0f);
      C[(size_t)m * ldc + n] = v;
      if (C16) C16[(size_t)m * ldc16 + n] = (h16)v;
    }
  }
}

// ---------------------------------------------------------------------------
// GRU step: gh = h_{t-1} @ Whh^T + bhh  (WMMA, K=128, N=384), then gate math.
// grid = B/16 blocks, 256 threads. Whole h tile staged by one TDM descriptor.
// ---------------------------------------------------------------------------
__global__ void __launch_bounds__(256) gru_step_kernel(
    const h16* __restrict__ hprev16, int ldh16, const h16* __restrict__ Whh,
    const float* __restrict__ gi, int ldgi, const float* __restrict__ bhh,
    const float* __restrict__ hprevf, int ldhf, float* __restrict__ hout,
    int ldho, h16* __restrict__ hout16, int ldho16) {
  __shared__ __align__(16) h16 sA[16 * 128];
  __shared__ float sGH[16 * 384];
  const int tid = threadIdx.x;
  const int lane = tid & 31;
  const int wave = tid >> 5;
  const int m0 = blockIdx.x * 16;

  if (hprev16) {
    stage_a_tile(sA, hprev16, (size_t)m0, ldh16, 0, 128, tid, wave);
  } else {
    for (int u = tid; u < 1024; u += 256) ((unsigned*)sA)[u] = 0u;
    __syncthreads();
  }

  v8f c0 = {}, c1 = {}, c2 = {};
  const int n0 = (wave * 3 + 0) * 16;
  const int n1 = (wave * 3 + 1) * 16;
  const int n2 = (wave * 3 + 2) * 16;
#pragma unroll
  for (int kkk = 0; kkk < 128; kkk += 32) {
    v16h a = load_a_frag_lds(sA, 128, kkk, lane);
    c0 = wmma_f32(a, load_b_frag(Whh, 128, n0, kkk, lane), c0);
    c1 = wmma_f32(a, load_b_frag(Whh, 128, n1, kkk, lane), c1);
    c2 = wmma_f32(a, load_b_frag(Whh, 128, n2, kkk, lane), c2);
  }
#pragma unroll
  for (int r = 0; r < 8; ++r) {
    const int row = r + ((lane < 16) ? 0 : 8);
    const int nn = lane & 15;
    sGH[row * 384 + n0 + nn] = c0[r];
    sGH[row * 384 + n1 + nn] = c1[r];
    sGH[row * 384 + n2 + nn] = c2[r];
  }
  __syncthreads();
  for (int idx = tid; idx < 16 * 128; idx += 256) {
    const int row = idx >> 7;
    const int col = idx & 127;
    const size_t grow = (size_t)(m0 + row);
    const float i_r = gi[grow * ldgi + col];
    const float i_z = gi[grow * ldgi + 128 + col];
    const float i_n = gi[grow * ldgi + 256 + col];
    const float g_r = sGH[row * 384 + col] + bhh[col];
    const float g_z = sGH[row * 384 + 128 + col] + bhh[128 + col];
    const float g_n = sGH[row * 384 + 256 + col] + bhh[256 + col];
    const float hp = hprevf ? hprevf[grow * ldhf + col] : 0.0f;
    const float rg = 1.0f / (1.0f + __expf(-(i_r + g_r)));
    const float zg = 1.0f / (1.0f + __expf(-(i_z + g_z)));
    const float ng = tanhf(i_n + rg * g_n);
    const float hv = (1.0f - zg) * ng + zg * hp;
    hout[grow * ldho + col] = hv;
    hout16[grow * ldho16 + col] = (h16)hv;
  }
}

// ---------------------------------------------------------------------------
// Fused CNN encoder per (b,t) image: conv3x3(SAME)+relu -> maxpool2 ->
// conv3x3(SAME)+relu -> avgpool2 -> feats (f16, K-padded to 608).
// ---------------------------------------------------------------------------
__global__ void __launch_bounds__(256) cnn_kernel(
    const float* __restrict__ veg, const float* __restrict__ cwsi,
    const float* __restrict__ vc1w, const float* __restrict__ vc1b,
    const float* __restrict__ vc2w, const float* __restrict__ vc2b,
    const float* __restrict__ cc1w, const float* __restrict__ cc1b,
    const float* __restrict__ cc2w, const float* __restrict__ cc2b,
    h16* __restrict__ feats16, int ldf) {
  __shared__ float s_w1[16 * 5 * 9];
  __shared__ float s_w2[16 * 16 * 9];
  __shared__ float s_cw1[16 * 1 * 9];
  __shared__ float s_cw2[8 * 16 * 9];
  __shared__ float s_b1[16], s_b2[16], s_cb1[16], s_cb2[8];
  __shared__ float s_in[5 * 22 * 22];   // reused as pooled [16*12*12]
  __shared__ float s_c1[16 * 20 * 20];  // reused as conv2 out [16*10*10]
  float* s_p = s_in;
  float* s_c2 = s_c1;

  const int tid = threadIdx.x;
  const int bt = blockIdx.x;

  for (int i = tid; i < 16 * 5 * 9; i += 256) s_w1[i] = vc1w[i];
  for (int i = tid; i < 16 * 16 * 9; i += 256) s_w2[i] = vc2w[i];
  for (int i = tid; i < 16 * 9; i += 256) s_cw1[i] = cc1w[i];
  for (int i = tid; i < 8 * 16 * 9; i += 256) s_cw2[i] = cc2w[i];
  if (tid < 16) { s_b1[tid] = vc1b[tid]; s_b2[tid] = vc2b[tid]; s_cb1[tid] = cc1b[tid]; }
  if (tid < 8) s_cb2[tid] = cc2b[tid];
  if (tid < 8) feats16[(size_t)bt * ldf + 600 + tid] = (h16)0.0f;  // K pad

#pragma unroll 1
  for (int half = 0; half < 2; ++half) {
    const int Cin = half ? 1 : 5;
    const int C2o = half ? 8 : 16;
    const float* img = half ? (cwsi + (size_t)bt * 400) : (veg + (size_t)bt * 5 * 400);
    const float* w1 = half ? s_cw1 : s_w1;
    const float* b1 = half ? s_cb1 : s_b1;
    const float* w2 = half ? s_cw2 : s_w2;
    const float* b2 = half ? s_cb2 : s_b2;
    const int foff = half ? 400 : 0;

    __syncthreads();
    for (int i = tid; i < Cin * 484; i += 256) {  // zero-padded 22x22 input
      const int ch = i / 484;
      const int p = i - ch * 484;
      const int y = p / 22, x = p % 22;
      float v = 0.0f;
      if (y >= 1 && y <= 20 && x >= 1 && x <= 20) v = img[ch * 400 + (y - 1) * 20 + (x - 1)];
      s_in[i] = v;
    }
    __syncthreads();
    for (int i = tid; i < 16 * 400; i += 256) {  // conv1 + relu
      const int oc = i / 400;
      const int p = i - oc * 400;
      const int y = p / 20, x = p % 20;
      float acc = b1[oc];
      for (int ic = 0; ic < Cin; ++ic) {
        const float* ip = s_in + ic * 484 + y * 22 + x;
        const float* wp = w1 + (oc * Cin + ic) * 9;
#pragma unroll
        for (int ky = 0; ky < 3; ++ky)
#pragma unroll
          for (int kx = 0; kx < 3; ++kx) acc = fmaf(ip[ky * 22 + kx], wp[ky * 3 + kx], acc);
      }
      s_c1[i] = fmaxf(acc, 0.0f);
    }
    __syncthreads();
    for (int i = tid; i < 16 * 144; i += 256) {  // maxpool2 -> zero-padded 12x12
      const int oc = i / 144;
      const int p = i - oc * 144;
      const int y = p / 12, x = p % 12;
      float v = 0.0f;
      if (y >= 1 && y <= 10 && x >= 1 && x <= 10) {
        const int sy = (y - 1) * 2, sx = (x - 1) * 2;
        const float* c1 = s_c1 + oc * 400;
        v = fmaxf(fmaxf(c1[sy * 20 + sx], c1[sy * 20 + sx + 1]),
                  fmaxf(c1[(sy + 1) * 20 + sx], c1[(sy + 1) * 20 + sx + 1]));
      }
      s_p[i] = v;
    }
    __syncthreads();
    for (int i = tid; i < C2o * 100; i += 256) {  // conv2 + relu
      const int oc = i / 100;
      const int p = i - oc * 100;
      const int y = p / 10, x = p % 10;
      float acc = b2[oc];
      for (int ic = 0; ic < 16; ++ic) {
        const float* ip = s_p + ic * 144 + y * 12 + x;
        const float* wp = w2 + (oc * 16 + ic) * 9;
#pragma unroll
        for (int ky = 0; ky < 3; ++ky)
#pragma unroll
          for (int kx = 0; kx < 3; ++kx) acc = fmaf(ip[ky * 12 + kx], wp[ky * 3 + kx], acc);
      }
      s_c2[i] = fmaxf(acc, 0.0f);
    }
    __syncthreads();
    for (int i = tid; i < C2o * 25; i += 256) {  // avgpool2 -> feats (f16)
      const int oc = i / 25;
      const int p = i - oc * 25;
      const int y = p / 5, x = p % 5;
      const float* c2 = s_c2 + oc * 100;
      const float v = 0.25f * (c2[(2 * y) * 10 + 2 * x] + c2[(2 * y) * 10 + 2 * x + 1] +
                               c2[(2 * y + 1) * 10 + 2 * x] + c2[(2 * y + 1) * 10 + 2 * x + 1]);
      feats16[(size_t)bt * ldf + foff + i] = (h16)v;
    }
  }
}

// ---------------------------------------------------------------------------
// Temporal attention + irrigation-embedding concat -> x[B][160] (144 + pad).
// ---------------------------------------------------------------------------
__global__ void __launch_bounds__(128) attn_kernel(
    const float* __restrict__ hseq, const float* __restrict__ w1,
    const float* __restrict__ b1, const float* __restrict__ w2,
    const float* __restrict__ b2, const float* __restrict__ emb,
    const int* __restrict__ irr, float* __restrict__ x, h16* __restrict__ x16,
    int ldx) {
  __shared__ float sh[10 * 128];
  __shared__ float shid[64];
  __shared__ float se[10];
  __shared__ float saw[10];
  const int b = blockIdx.x, tid = threadIdx.x;
  for (int i = tid; i < 1280; i += 128) sh[i] = hseq[(size_t)b * 1280 + i];
  __syncthreads();
  for (int t = 0; t < 10; ++t) {
    if (tid < 64) {
      float acc = b1[tid];
      const float* wr = w1 + tid * 128;
      for (int k = 0; k < 128; ++k) acc = fmaf(sh[t * 128 + k], wr[k], acc);
      shid[tid] = tanhf(acc);
    }
    __syncthreads();
    if (tid == 0) {
      float e = b2[0];
      for (int j = 0; j < 64; ++j) e += shid[j] * w2[j];
      se[t] = e;
    }
    __syncthreads();
  }
  if (tid == 0) {
    float mx = se[0];
    for (int t = 1; t < 10; ++t) mx = fmaxf(mx, se[t]);
    float s = 0.0f;
    for (int t = 0; t < 10; ++t) { saw[t] = __expf(se[t] - mx); s += saw[t]; }
    for (int t = 0; t < 10; ++t) saw[t] /= s;
  }
  __syncthreads();
  {
    float acc = 0.0f;
    for (int t = 0; t < 10; ++t) acc = fmaf(saw[t], sh[t * 128 + tid], acc);
    x[(size_t)b * ldx + tid] = acc;
    x16[(size_t)b * ldx + tid] = (h16)acc;
  }
  if (tid < 32) {
    const int col = 128 + tid;
    const float v = (tid < 16) ? emb[irr[b] * 16 + tid] : 0.0f;
    x[(size_t)b * ldx + col] = v;
    x16[(size_t)b * ldx + col] = (h16)v;
  }
}

// ---------------------------------------------------------------------------
// segment_max via monotonic-uint atomicMax
// ---------------------------------------------------------------------------
static __device__ __forceinline__ unsigned fenc(float f) {
  unsigned u = __float_as_uint(f);
  return (u & 0x80000000u) ? ~u : (u | 0x80000000u);
}
static __device__ __forceinline__ float fdec(unsigned e) {
  unsigned u = (e & 0x80000000u) ? (e & 0x7FFFFFFFu) : ~e;
  return __uint_as_float(u);
}

__global__ void fill_kernel(unsigned* p, unsigned v, int n) {
  const int i = blockIdx.x * blockDim.x + threadIdx.x;
  if (i < n) p[i] = v;
}

__global__ void segmax_kernel(const float* __restrict__ x, int ldx,
                              const int* __restrict__ src, const int* __restrict__ dst,
                              int E, int D, unsigned* __restrict__ agg, int lda) {
  const int i = blockIdx.x * blockDim.x + threadIdx.x;
  if (i >= E * D) return;
  const int e = i / D, d = i - e * D;
  const float v = x[(size_t)src[e] * ldx + d];
  atomicMax(&agg[(size_t)dst[e] * lda + d], fenc(v));
}

__global__ void decode_kernel(const unsigned* __restrict__ agg, h16* __restrict__ out16,
                              int n, int D, int Dp) {
  const int i = blockIdx.x * blockDim.x + threadIdx.x;
  if (i >= n * Dp) return;
  const int d = i % Dp;
  float v = 0.0f;
  if (d < D) {
    const float f = fdec(agg[i]);
    if (f > -1e30f && f < 1e30f) v = f;  // empty neighborhood (-inf) -> 0
  }
  out16[i] = (h16)v;
}

__global__ void pack_kernel(const float* __restrict__ src, h16* __restrict__ dst,
                            int N, int K, int Kp) {
  const int i = blockIdx.x * blockDim.x + threadIdx.x;
  if (i >= N * Kp) return;
  const int n = i / Kp, k = i - n * Kp;
  dst[i] = (h16)((k < K) ? src[(size_t)n * K + k] : 0.0f);
}

__global__ void final_kernel(const float* __restrict__ x2, const float* __restrict__ fw,
                             const float* __restrict__ fb, float* __restrict__ out, int n) {
  const int i = blockIdx.x * blockDim.x + threadIdx.x;
  if (i >= n) return;
  float acc = fb[0];
  for (int j = 0; j < 32; ++j) acc = fmaf(x2[(size_t)i * 32 + j], fw[j], acc);
  out[i] = acc;
}

// ---------------------------------------------------------------------------
extern "C" void kernel_launch(void* const* d_in, const int* in_sizes, int n_in,
                              void* d_out, int out_size, void* d_ws, size_t ws_size,
                              hipStream_t stream) {
  (void)in_sizes; (void)n_in; (void)out_size; (void)ws_size;
  const float* veg = (const float*)d_in[0];
  const float* cwsi = (const float*)d_in[1];
  const int* irr = (const int*)d_in[2];
  const int* eidx = (const int*)d_in[3];
  const float* vc1w = (const float*)d_in[4]; const float* vc1b = (const float*)d_in[5];
  const float* vc2w = (const float*)d_in[6]; const float* vc2b = (const float*)d_in[7];
  const float* cc1w = (const float*)d_in[8]; const float* cc1b = (const float*)d_in[9];
  const float* cc2w = (const float*)d_in[10]; const float* cc2b = (const float*)d_in[11];
  const float* emb = (const float*)d_in[12];
  const float* g0wih = (const float*)d_in[13]; const float* g0whh = (const float*)d_in[14];
  const float* g0bih = (const float*)d_in[15]; const float* g0bhh = (const float*)d_in[16];
  const float* g1wih = (const float*)d_in[17]; const float* g1whh = (const float*)d_in[18];
  const float* g1bih = (const float*)d_in[19]; const float* g1bhh = (const float*)d_in[20];
  const float* a1w = (const float*)d_in[21]; const float* a1b = (const float*)d_in[22];
  const float* a2w = (const float*)d_in[23]; const float* a2b = (const float*)d_in[24];
  const float* s1wl = (const float*)d_in[25]; const float* s1bl = (const float*)d_in[26];
  const float* s1wr = (const float*)d_in[27];
  const float* s2wl = (const float*)d_in[28]; const float* s2bl = (const float*)d_in[29];
  const float* s2wr = (const float*)d_in[30];
  const float* fcw = (const float*)d_in[31]; const float* fcb = (const float*)d_in[32];
  float* out = (float*)d_out;

  const int B = 2048, T = 10, BT = B * T, E = 32768;
  const int KF = 608;  // 600 padded to multiple of 32
  const int H = 128, G = 384;

  char* base = (char*)d_ws;
  size_t off = 0;
  auto alloc = [&](size_t bytes) -> char* {
    off = (off + 255) & ~(size_t)255;
    char* p = base + off;
    off += bytes;
    return p;
  };

  h16* feats16 = (h16*)alloc((size_t)BT * KF * 2);
  h16* w0ih16 = (h16*)alloc((size_t)G * KF * 2);
  h16* w0hh16 = (h16*)alloc((size_t)G * H * 2);
  h16* w1ih16 = (h16*)alloc((size_t)G * H * 2);
  h16* w1hh16 = (h16*)alloc((size_t)G * H * 2);
  h16* s1wl16 = (h16*)alloc((size_t)64 * 160 * 2);
  h16* s1wr16 = (h16*)alloc((size_t)64 * 160 * 2);
  h16* s2wl16 = (h16*)alloc((size_t)32 * 64 * 2);
  h16* s2wr16 = (h16*)alloc((size_t)32 * 64 * 2);
  float* gi0 = (float*)alloc((size_t)BT * G * 4);
  float* gi1 = (float*)alloc((size_t)BT * G * 4);
  float* hs0 = (float*)alloc((size_t)BT * H * 4);
  h16* hs0h = (h16*)alloc((size_t)BT * H * 2);
  float* hs1 = (float*)alloc((size_t)BT * H * 4);
  h16* hs1h = (h16*)alloc((size_t)BT * H * 2);
  float* xf = (float*)alloc((size_t)B * 160 * 4);
  h16* x16 = (h16*)alloc((size_t)B * 160 * 2);
  unsigned* agg1 = (unsigned*)alloc((size_t)B * 160 * 4);
  h16* agg1h = (h16*)alloc((size_t)B * 160 * 2);
  float* x1f = (float*)alloc((size_t)B * 64 * 4);
  h16* x1h = (h16*)alloc((size_t)B * 64 * 2);
  unsigned* agg2 = (unsigned*)alloc((size_t)B * 64 * 4);
  h16* agg2h = (h16*)alloc((size_t)B * 64 * 2);
  float* x2f = (float*)alloc((size_t)B * 32 * 4);

  // 1. pack weights to K-padded f16
  auto pk = [&](const float* s, h16* d, int N_, int K_, int Kp_) {
    const int n = N_ * Kp_;
    pack_kernel<<<(n + 255) / 256, 256, 0, stream>>>(s, d, N_, K_, Kp_);
  };
  pk(g0wih, w0ih16, G, 600, KF);
  pk(g0whh, w0hh16, G, H, H);
  pk(g1wih, w1ih16, G, H, H);
  pk(g1whh, w1hh16, G, H, H);
  pk(s1wl, s1wl16, 64, 144, 160);
  pk(s1wr, s1wr16, 64, 144, 160);
  pk(s2wl, s2wl16, 32, 64, 64);
  pk(s2wr, s2wr16, 32, 64, 64);

  // 2. CNN encoders -> feats16 [BT x 608]
  cnn_kernel<<<BT, 256, 0, stream>>>(veg, cwsi, vc1w, vc1b, vc2w, vc2b, cc1w,
                                     cc1b, cc2w, cc2b, feats16, KF);

  // 3. gi0 = feats @ W0ih^T + b0ih
  gemm_wmma_kernel<<<dim3(BT / 16, 3), 256, 0, stream>>>(
      feats16, KF, w0ih16, KF, nullptr, 0, nullptr, 0, g0bih, gi0, G, nullptr,
      0, BT, G, 0);

  // 4. GRU layer 0 (sequential over T)
  for (int t = 0; t < T; ++t) {
    gru_step_kernel<<<B / 16, 256, 0, stream>>>(
        t ? hs0h + (size_t)(t - 1) * H : nullptr, T * H, w0hh16,
        gi0 + (size_t)t * G, T * G, g0bhh,
        t ? hs0 + (size_t)(t - 1) * H : nullptr, T * H,
        hs0 + (size_t)t * H, T * H, hs0h + (size_t)t * H, T * H);
  }

  // 5. gi1 = h0seq @ W1ih^T + b1ih
  gemm_wmma_kernel<<<dim3(BT / 16, 3), 256, 0, stream>>>(
      hs0h, H, w1ih16, H, nullptr, 0, nullptr, 0, g1bih, gi1, G, nullptr, 0,
      BT, G, 0);

  // 6. GRU layer 1
  for (int t = 0; t < T; ++t) {
    gru_step_kernel<<<B / 16, 256, 0, stream>>>(
        t ? hs1h + (size_t)(t - 1) * H : nullptr, T * H, w1hh16,
        gi1 + (size_t)t * G, T * G, g1bhh,
        t ? hs1 + (size_t)(t - 1) * H : nullptr, T * H,
        hs1 + (size_t)t * H, T * H, hs1h + (size_t)t * H, T * H);
  }

  // 7. attention + embedding concat -> x [B x 160] (144 valid)
  attn_kernel<<<B, 128, 0, stream>>>(hs1, a1w, a1b, a2w, a2b, emb, irr, xf, x16, 160);

  // 8. SAGE layer 1: x1 = relu(agg@Wl^T + bl + x@Wr^T)
  fill_kernel<<<(B * 160 + 255) / 256, 256, 0, stream>>>(agg1, 0x007FFFFFu, B * 160);
  segmax_kernel<<<(E * 144 + 255) / 256, 256, 0, stream>>>(xf, 160, eidx, eidx + E, E, 144, agg1, 160);
  decode_kernel<<<(B * 160 + 255) / 256, 256, 0, stream>>>(agg1, agg1h, B, 144, 160);
  gemm_wmma_kernel<<<dim3(B / 16, 1), 256, 0, stream>>>(
      agg1h, 160, s1wl16, 160, x16, 160, s1wr16, 160, s1bl, x1f, 64, x1h, 64,
      B, 64, 1);

  // 9. SAGE layer 2
  fill_kernel<<<(B * 64 + 255) / 256, 256, 0, stream>>>(agg2, 0x007FFFFFu, B * 64);
  segmax_kernel<<<(E * 64 + 255) / 256, 256, 0, stream>>>(x1f, 64, eidx, eidx + E, E, 64, agg2, 64);
  decode_kernel<<<(B * 64 + 255) / 256, 256, 0, stream>>>(agg2, agg2h, B, 64, 64);
  gemm_wmma_kernel<<<dim3(B / 16, 1), 256, 0, stream>>>(
      agg2h, 64, s2wl16, 64, x1h, 64, s2wr16, 64, s2bl, x2f, 32, nullptr, 0,
      B, 32, 1);

  // 10. final FC -> out [B x 1]
  final_kernel<<<(B + 255) / 256, 256, 0, stream>>>(x2f, fcw, fcb, out, B);
}
